// PitchRegulator_44727789421052
// MI455X (gfx1250) — matrix operand order repaired and checked by reading, not compiled, not run on hardware
//
#include <hip/hip_runtime.h>
#include <stdint.h>

// ---------------- problem constants ----------------
#define kB 16
#define kT 2048
#define kC 256
#define kF 256
#define kK 768            // 3 taps * 256
#define kW 10
#define MT 32             // output rows per workgroup in the conv GEMMs
#define XS_STRIDE 264     // padded LDS row stride (bf16 elems) to avoid bank conflicts
#define PI_F 3.14159265358979323846f

typedef __bf16 bf16_t;
typedef bf16_t bf16x8 __attribute__((ext_vector_type(8)));
typedef bf16_t v16bf  __attribute__((ext_vector_type(16)));
typedef float  v8f    __attribute__((ext_vector_type(8)));

// ---------------- small utility kernels ----------------
__global__ void zero_n_kernel(float* p, int n) {
  int i = blockIdx.x * blockDim.x + threadIdx.x;
  if (i < n) p[i] = 0.0f;
}

__global__ void cvt_bf16_kernel(const float* __restrict__ in, bf16_t* __restrict__ out, int n) {
  int i = blockIdx.x * blockDim.x + threadIdx.x;
  if (i < n) out[i] = (bf16_t)in[i];
}

// w: (F,C,3) OIW  ->  wp[f*768 + k*256 + c]  (B^T layout: row f, col kc)
__global__ void prep_w_kernel(const float* __restrict__ w, bf16_t* __restrict__ wp) {
  int i = blockIdx.x * blockDim.x + threadIdx.x;
  if (i < kF * kK) {
    int f = i / kK;
    int kc = i % kK;
    int k = kc >> 8;
    int c = kc & 255;
    wp[i] = (bf16_t)w[(size_t)f * (kC * 3) + c * 3 + k];
  }
}

// ---------------- per-batch target stats + normalized pitch contour ----------------
__global__ __launch_bounds__(256) void stats_cnorm_kernel(
    const float* __restrict__ target, const int* __restrict__ mel_len,
    float* __restrict__ cnorm, float* __restrict__ out_tparams) {
  __shared__ float red[256];
  const int b = blockIdx.x, tid = threadIdx.x;
  const float* tb = target + (size_t)b * kT;

  float s = 0.0f;
  for (int t = tid; t < kT; t += 256) s += tb[t];
  red[tid] = s;
  __syncthreads();
  for (int off = 128; off > 0; off >>= 1) {
    if (tid < off) red[tid] += red[tid + off];
    __syncthreads();
  }
  const int len = mel_len[b];
  const float lensf = (float)len;
  const float mean = red[0] / lensf;
  __syncthreads();

  float s2 = 0.0f;
  for (int t = tid; t < kT; t += 256) {
    float d = (t < len) ? (tb[t] - mean) : 0.0f;
    s2 += d * d;
  }
  red[tid] = s2;
  __syncthreads();
  for (int off = 128; off > 0; off >>= 1) {
    if (tid < off) red[tid] += red[tid + off];
    __syncthreads();
  }
  const float stdv = sqrtf(red[0] / (lensf - 1.0f));
  if (tid == 0) {
    out_tparams[b * 2 + 0] = mean;
    out_tparams[b * 2 + 1] = stdv;
  }
  const float inv = 1.0f / stdv;
  for (int t = tid; t < kT; t += 256) {
    float d = (t < len) ? (tb[t] - mean) : 0.0f;
    cnorm[(size_t)b * kT + t] = d * inv;
  }
}

// ---------------- time-domain Morlet filters: psi = ifft(psi_hat) ----------------
// psi_hat real, nonzero only for freq bins j = 1..1023 (omega>0).
__global__ __launch_bounds__(256) void psi_prep_kernel(float* __restrict__ psiRe,
                                                       float* __restrict__ psiIm) {
  const int sidx = blockIdx.x;                  // scale 0..9
  const float s = expf(0.4f * (float)sidx);     // SCALES
  const float amp = 0.7511255444649425f * sqrtf(2.0f * PI_F * s);  // pi^-0.25 * sqrt(2*pi*s)
  const float w0 = 6.0f;
  for (int tau = threadIdx.x; tau < kT; tau += 256) {
    float re = 0.0f, im = 0.0f;
    for (int j = 1; j < kT / 2; ++j) {
      const float omega = 2.0f * PI_F * (float)j / (float)kT;
      const float d = s * omega - w0;
      const float ph = amp * expf(-0.5f * d * d);
      const int r = (j * tau) & (kT - 1);       // exact angle reduction
      const float ang = (2.0f * PI_F / (float)kT) * (float)r;
      float sn, cs;
      __sincosf(ang, &sn, &cs);
      re += ph * cs;
      im += ph * sn;
    }
    psiRe[sidx * kT + tau] = re * (1.0f / (float)kT);
    psiIm[sidx * kT + tau] = im * (1.0f / (float)kT);
  }
}

// ---------------- CWT = circular convolution with LDS-staged operands ----------------
__global__ __launch_bounds__(256) void cwt_kernel(
    const float* __restrict__ cnorm, const float* __restrict__ psiRe,
    const float* __restrict__ psiIm, const int* __restrict__ mel_len,
    float* __restrict__ out_tspec) {
  __shared__ float cs_[kT];
  __shared__ float pr[kT];
  __shared__ float pim[kT];
  const int chunks = kT / 256;                  // 8
  const int blk = blockIdx.x;
  const int b = blk / (kW * chunks);
  const int rem = blk % (kW * chunks);
  const int sc = rem / chunks;
  const int tc = rem % chunks;
  for (int i = threadIdx.x; i < kT; i += 256) {
    cs_[i] = cnorm[(size_t)b * kT + i];
    pr[i] = psiRe[sc * kT + i];
    pim[i] = psiIm[sc * kT + i];
  }
  __syncthreads();
  const int t = tc * 256 + threadIdx.x;
  float re = 0.0f, im = 0.0f;
  for (int tau = 0; tau < kT; ++tau) {
    const int idx = (t - tau) & (kT - 1);
    const float c = cs_[tau];
    re += c * pr[idx];
    im += c * pim[idx];
  }
  const float msk = (t < mel_len[b]) ? 1.0f : 0.0f;
  const size_t base = ((size_t)b * kT + t) * 20;
  out_tspec[base + sc] = re * msk;
  out_tspec[base + 10 + sc] = im * msk;
}

// ---------------- embedding gather ----------------
__global__ __launch_bounds__(64) void embed_kernel(
    const float* __restrict__ target, const float* __restrict__ alpha,
    const int* __restrict__ mel_len, const float* __restrict__ emb,
    float* __restrict__ out) {
  const int bt = blockIdx.x;
  const int b = bt >> 11;
  const int t = bt & (kT - 1);
  float q = target[(size_t)b * kT + t] * alpha[b];
  q = fminf(fmaxf(q, 0.0f), 800.0f);
  int qi = (int)rintf(q * (1.0f / 3.125f)) + 1;   // RNE matches jnp.round
  if (t >= mel_len[b]) qi = 0;
  const float4* src = reinterpret_cast<const float4*>(emb + (size_t)qi * 256);
  float4* dst = reinterpret_cast<float4*>(out + (size_t)bt * 256);
  dst[threadIdx.x] = src[threadIdx.x];
}

// ---------------- fused conv1d(k=3) GEMM + bias + LN + ReLU (+ linear head) ----------
// y[b,t,f] = sum_{k,c} A[b,t-1+k,c] * w[f,c,k];  Wp layout: [f][k*256+c] (B^T).
// 8 waves: 2 (M) x 4 (N); each wave: one 16-row A tile x four 16-col B tiles,
// K = 768 consumed as 24 steps of v_wmma_f32_16x16x32_bf16.
template <int STAGE>
__global__ __launch_bounds__(256) void conv_gemm_kernel(
    const bf16_t* __restrict__ A, const bf16_t* __restrict__ Wp,
    const float* __restrict__ cbias, const float* __restrict__ lng,
    const float* __restrict__ lnb, bf16_t* __restrict__ Hout,
    const float* __restrict__ lin_w, const float* __restrict__ lin_b,
    const int* __restrict__ mel_len, float* __restrict__ out_spec,
    float* __restrict__ out_params) {
  __shared__ bf16_t xs[(MT + 2) * XS_STRIDE];   // conv input tile (rows t0-1..t0+MT)
  __shared__ float hs[MT][kC + 1];              // f32 output tile (padded)
  __shared__ float rmean[MT];
  __shared__ float rrstd[MT];

  const int tid = threadIdx.x;
  const int blk = blockIdx.x;
  const int b = blk >> 6;                       // 64 tiles of MT=32 per batch
  const int t0 = (blk & 63) * MT;

  // ---- stage input rows into LDS (zero pad outside [0,T)) ----
  for (int chunk = tid; chunk < (MT + 2) * (kC / 8); chunk += 256) {
    const int r = chunk / (kC / 8);
    const int c8 = chunk % (kC / 8);
    const int t = t0 - 1 + r;
    bf16x8 v;
    if (t >= 0 && t < kT) {
      v = *reinterpret_cast<const bf16x8*>(A + ((size_t)b * kT + t) * kC + c8 * 8);
    } else {
#pragma unroll
      for (int j = 0; j < 8; ++j) v[j] = (bf16_t)0.0f;
    }
    *reinterpret_cast<bf16x8*>(&xs[r * XS_STRIDE + c8 * 8]) = v;
  }
  __syncthreads();

  const int lane = tid & 31;
  const int wid = tid >> 5;
  const int wm = wid >> 2;                      // 0..1 -> which 16-row tile
  const int wn = wid & 3;                       // 0..3 -> which 64-col group
  const int m16 = lane & 15;
  const int kh = lane >> 4;

  v8f acc[4];
#pragma unroll
  for (int ni = 0; ni < 4; ++ni)
#pragma unroll
    for (int e = 0; e < 8; ++e) acc[ni][e] = 0.0f;

  const int arow = wm * 16 + m16;

  for (int ktap = 0; ktap < 3; ++ktap) {
    // prefetch this tap's weight stream (global_prefetch_b8 on gfx1250)
    __builtin_prefetch(Wp + (size_t)(wn * 64 + m16) * kK + ktap * kC, 0, 1);
    for (int c0 = 0; c0 < kC; c0 += 32) {
      const int kc0 = ktap * kC + c0;
      // A fragment (16x32 bf16 layout per ISA: lane m16, halves kh; two 16B LDS reads)
      const bf16_t* ap = &xs[(arow + ktap) * XS_STRIDE + c0 + kh * 8];
      bf16x8 alo = *reinterpret_cast<const bf16x8*>(ap);
      bf16x8 ahi = *reinterpret_cast<const bf16x8*>(ap + 16);
      v16bf afrag;
#pragma unroll
      for (int j = 0; j < 8; ++j) {
        afrag[j] = alo[j];
        afrag[j + 8] = ahi[j];
      }
#pragma unroll
      for (int ni = 0; ni < 4; ++ni) {
        // B fragment: lane n=m16, K runs kh*8..+7 and 16+kh*8..+7 in Wp row f
        const int f = wn * 64 + ni * 16 + m16;
        const bf16_t* bp = Wp + (size_t)f * kK + kc0 + kh * 8;
        bf16x8 blo = *reinterpret_cast<const bf16x8*>(bp);
        bf16x8 bhi = *reinterpret_cast<const bf16x8*>(bp + 16);
        v16bf bfrag;
#pragma unroll
        for (int j = 0; j < 8; ++j) {
          bfrag[j] = blo[j];
          bfrag[j + 8] = bhi[j];
        }
        acc[ni] = __builtin_amdgcn_wmma_f32_16x16x32_bf16(
            false, afrag, false, bfrag, (short)0, acc[ni], false, false);
      }
    }
  }

  // ---- spill accumulators (+conv bias) to LDS: D layout m = kh*8+e, n = m16 ----
#pragma unroll
  for (int ni = 0; ni < 4; ++ni) {
    const int n = wn * 64 + ni * 16 + m16;
    const float bb = cbias[n];
#pragma unroll
    for (int e = 0; e < 8; ++e) {
      const int m = wm * 16 + kh * 8 + e;
      hs[m][n] = acc[ni][e] + bb;
    }
  }
  __syncthreads();

  // ---- LayerNorm stats per row ----
  if (tid < MT) {
    float s = 0.0f;
    for (int f = 0; f < kC; ++f) s += hs[tid][f];
    const float mean = s * (1.0f / kC);
    float v = 0.0f;
    for (int f = 0; f < kC; ++f) {
      const float d = hs[tid][f] - mean;
      v += d * d;
    }
    rmean[tid] = mean;
    rrstd[tid] = rsqrtf(v * (1.0f / kC) + 1e-5f);
  }
  __syncthreads();

  // ---- normalize + ReLU; stage1 -> bf16 h1 to HBM, stage2 -> keep in LDS ----
  for (int idx = tid; idx < MT * kC; idx += 256) {
    const int m = idx >> 8;
    const int f = idx & 255;
    float val = (hs[m][f] - rmean[m]) * rrstd[m] * lng[f] + lnb[f];
    val = fmaxf(val, 0.0f);
    if (STAGE == 1) {
      Hout[((size_t)b * kT + t0 + m) * kC + f] = (bf16_t)val;
    } else {
      hs[m][f] = val;
    }
  }

  if (STAGE == 2) {
    __syncthreads();
    const int len = mel_len[b];
    for (int task = tid; task < MT * 22; task += 256) {
      const int m = task / 22;
      const int o = task % 22;
      float s = lin_b[o];
      for (int f = 0; f < kC; ++f) s += hs[m][f] * lin_w[o * kC + f];
      const int t = t0 + m;
      if (o < 20) {
        out_spec[((size_t)b * kT + t) * 20 + o] = (t < len) ? s : 0.0f;
      } else {
        atomicAdd(&out_params[b * 2 + (o - 20)], s * (1.0f / (float)kT));
      }
    }
  }
}

// ---------------- launcher ----------------
extern "C" void kernel_launch(void* const* d_in, const int* in_sizes, int n_in,
                              void* d_out, int out_size, void* d_ws, size_t ws_size,
                              hipStream_t stream) {
  (void)in_sizes; (void)n_in; (void)out_size; (void)ws_size;

  const float* x       = (const float*)d_in[0];
  const float* alpha   = (const float*)d_in[1];
  const float* target  = (const float*)d_in[2];
  const int*   mel_len = (const int*)d_in[3];
  const float* conv1_w = (const float*)d_in[4];
  const float* conv1_b = (const float*)d_in[5];
  const float* ln1_g   = (const float*)d_in[6];
  const float* ln1_b   = (const float*)d_in[7];
  const float* conv2_w = (const float*)d_in[8];
  const float* conv2_b = (const float*)d_in[9];
  const float* ln2_g   = (const float*)d_in[10];
  const float* ln2_b   = (const float*)d_in[11];
  const float* lin_w   = (const float*)d_in[12];
  const float* lin_b   = (const float*)d_in[13];
  const float* emb     = (const float*)d_in[14];
  float* out = (float*)d_out;

  // output regions (flat concat in reference return order)
  float* out_emb     = out;                              // 16*2048*256
  float* out_pspec   = out + (size_t)8388608;            // 16*2048*20
  float* out_pparams = out + (size_t)9043968;            // 16*2
  float* out_tspec   = out + (size_t)9044000;            // 16*2048*20
  float* out_tparams = out + (size_t)9699360;            // 16*2

  // workspace carve-out
  uintptr_t base = (uintptr_t)d_ws;
  size_t off = 0;
  auto take = [&](size_t bytes) -> void* {
    off = (off + 255) & ~(size_t)255;
    void* p = (void*)(base + off);
    off += bytes;
    return p;
  };
  bf16_t* xb    = (bf16_t*)take((size_t)kB * kT * kC * sizeof(bf16_t));
  bf16_t* h1    = (bf16_t*)take((size_t)kB * kT * kC * sizeof(bf16_t));
  bf16_t* w1p   = (bf16_t*)take((size_t)kF * kK * sizeof(bf16_t));
  bf16_t* w2p   = (bf16_t*)take((size_t)kF * kK * sizeof(bf16_t));
  float*  cnorm = (float*)take((size_t)kB * kT * sizeof(float));
  float*  psiRe = (float*)take((size_t)kW * kT * sizeof(float));
  float*  psiIm = (float*)take((size_t)kW * kT * sizeof(float));

  // predictor_params is accumulated atomically -> zero it first
  zero_n_kernel<<<1, 32, 0, stream>>>(out_pparams, 32);

  // bf16 conversions / weight transposes
  {
    const int n = kB * kT * kC;
    cvt_bf16_kernel<<<n / 256, 256, 0, stream>>>(x, xb, n);
  }
  prep_w_kernel<<<(kF * kK) / 256, 256, 0, stream>>>(conv1_w, w1p);
  prep_w_kernel<<<(kF * kK) / 256, 256, 0, stream>>>(conv2_w, w2p);

  // target stats + normalized contour; Morlet filters
  stats_cnorm_kernel<<<kB, 256, 0, stream>>>(target, mel_len, cnorm, out_tparams);
  psi_prep_kernel<<<kW, 256, 0, stream>>>(psiRe, psiIm);

  // conv1 -> h1 (bf16), conv2 + linear head fused
  conv_gemm_kernel<1><<<(kB * kT) / MT, 256, 0, stream>>>(
      xb, w1p, conv1_b, ln1_g, ln1_b, h1,
      nullptr, nullptr, nullptr, nullptr, nullptr);
  conv_gemm_kernel<2><<<(kB * kT) / MT, 256, 0, stream>>>(
      h1, w2p, conv2_b, ln2_g, ln2_b, nullptr,
      lin_w, lin_b, mel_len, out_pspec, out_pparams);

  // CWT target spectrogram
  cwt_kernel<<<kB * kW * (kT / 256), 256, 0, stream>>>(cnorm, psiRe, psiIm, mel_len,
                                                       out_tspec);

  // embedding gather
  embed_kernel<<<kB * kT, 64, 0, stream>>>(target, alpha, mel_len, emb, out_emb);
}
